// GQAAttention_77799037599799
// MI455X (gfx1250) — matrix operand order repaired and checked by reading, compiled, not run on hardware
//
#include <hip/hip_runtime.h>
#include <math.h>

typedef __attribute__((ext_vector_type(16))) __bf16 v16bf;
typedef __attribute__((ext_vector_type(8)))  float  v8f;
typedef __attribute__((ext_vector_type(4)))  unsigned int v4u;
typedef __attribute__((ext_vector_type(8)))  int v8i;
typedef __attribute__((ext_vector_type(4)))  int v4i;

union ABFrag { v16bf v; unsigned u[8]; };

__device__ __forceinline__ unsigned pack_bf16(float a, float b) {
    __bf16 lo = (__bf16)a, hi = (__bf16)b;
    unsigned short ul = __builtin_bit_cast(unsigned short, lo);
    unsigned short uh = __builtin_bit_cast(unsigned short, hi);
    return ((unsigned)uh << 16) | (unsigned)ul;
}

#define D_MODEL   2048
#define N_HEADS   32
#define N_KV      8
#define HDIM      64
#define SEQ       2048
#define BTOK      4096   // B * S
#define BM        128
#define BN        128
#define BK        32

// ---------------------------------------------------------------------------
// TDM: 2D tile load Global -> LDS via Tensor Data Mover.
// rows x cols(bf16) tile, row stride in elements; descriptor per ISA 08 §8.
// 6-arg builtin form (clang-23 / therock-10.0 toolchain).
// ---------------------------------------------------------------------------
__device__ __forceinline__ void tdm_load_tile_bf16(
    unsigned lds_byte_off, const __bf16* gptr,
    unsigned tile_cols, unsigned tile_rows, unsigned row_stride_elems)
{
    unsigned long long ga = (unsigned long long)(uintptr_t)gptr;
    v4u g0;
    g0[0] = 1u;                                        // count=1 (valid), load, user
    g0[1] = lds_byte_off;                              // lds_addr (bytes)
    g0[2] = (unsigned)(ga & 0xFFFFFFFFu);              // global_addr[31:0]
    g0[3] = (unsigned)((ga >> 32) & 0x01FFFFFFu)       // global_addr[56:32]
          | (2u << 30);                                // type = 2 (image)
    v8i g1;
    g1[0] = (int)(1u << 16);                           // data_size = 2 bytes (code 1)
    g1[1] = (int)(tile_cols << 16);                    // tensor_dim0 lo16 in [31:16]
    g1[2] = (int)(tile_rows << 16);                    // tensor_dim0 hi=0; tensor_dim1 lo in [31:16]
    g1[3] = (int)(tile_cols << 16);                    // tensor_dim1 hi=0; tile_dim0 in [31:16]
    g1[4] = (int)tile_rows;                            // tile_dim1 in [15:0]; tile_dim2=0
    g1[5] = (int)row_stride_elems;                     // tensor_dim0_stride[31:0]
    g1[6] = 0;                                         // stride hi, tensor_dim1_stride lo
    g1[7] = 0;
    v4i z4 = {0, 0, 0, 0};
    v8i z8 = {0, 0, 0, 0, 0, 0, 0, 0};
    __builtin_amdgcn_tensor_load_to_lds(g0, g1, z4, z4, z8, 0);
}

// ---------------------------------------------------------------------------
// Kernel 1: fused QKV projection. C = x @ [Wq | Wk | Wv], bf16 WMMA, f32 acc.
// grid = (BTOK/BM, 3072/BN) = (32, 24), block = 256 (8 waves, 4Mx2N wave grid)
// per-wave tile 32x64 -> 8 WMMAs per K-step from 6 fragment loads
// ---------------------------------------------------------------------------
__global__ __launch_bounds__(256) void qkv_gemm_kernel(
    const float* __restrict__ x,
    const float* __restrict__ Wq,
    const float* __restrict__ Wk,
    const float* __restrict__ Wv,
    __bf16* __restrict__ qout,   // [BTOK, 2048]
    __bf16* __restrict__ kout,   // [BTOK, 512]
    __bf16* __restrict__ vout)   // [BTOK, 512]
{
    __shared__ unsigned lds_a[BM][BK/2];   // bf16 pairs [m][k/2]  (8 KB)
    __shared__ unsigned lds_b[BN][BK/2];   // transposed [n][k/2]  (8 KB)

    const int tid  = threadIdx.x;
    const int wave = tid >> 5;
    const int lane = tid & 31;
    const int wm   = wave >> 1;            // 0..3 -> m offset wm*32
    const int wn   = wave & 1;             // 0..1 -> n offset wn*64

    const int row0     = blockIdx.x * BM;
    const int n_global = blockIdx.y * BN;

    const float* W; int ldw; __bf16* out; int ldo; int ncol0;
    if (n_global < 2048)      { W = Wq; ldw = 2048; out = qout; ldo = 2048; ncol0 = n_global; }
    else if (n_global < 2560) { W = Wk; ldw = 512;  out = kout; ldo = 512;  ncol0 = n_global - 2048; }
    else                      { W = Wv; ldw = 512;  out = vout; ldo = 512;  ncol0 = n_global - 2560; }

    v8f acc[2][4] = {};   // [mi 16][ni 16]

    for (int k0 = 0; k0 < D_MODEL; k0 += BK) {
        // A tile: 128x32 f32 -> bf16 pairs (2048 uints, 8 per thread)
        for (int i = 0; i < 8; ++i) {
            int u = tid * 8 + i;
            int m = u >> 4, kp = u & 15;
            const float* p = x + (size_t)(row0 + m) * D_MODEL + k0 + kp * 2;
            lds_a[m][kp] = pack_bf16(p[0], p[1]);
            __builtin_prefetch(p + BK, 0, 3);   // speculative: OOB dropped silently
        }
        // B tile transposed: W[k][n] -> lds_b[n][k/2] (2048 uints, 8 per thread)
        for (int i = 0; i < 8; ++i) {
            int u = tid * 8 + i;
            int n = u >> 4, kp = u & 15;
            const float* p = W + (size_t)(k0 + kp * 2) * ldw + ncol0 + n;
            lds_b[n][kp] = pack_bf16(p[0], p[ldw]);
            __builtin_prefetch(p + (size_t)BK * ldw, 0, 3);
        }
        __syncthreads();

        ABFrag afr[2], bfr[4];
        for (int mi = 0; mi < 2; ++mi) {
            int m = wm * 32 + mi * 16 + (lane & 15);
            for (int r = 0; r < 8; ++r) {
                int kk = ((r >> 2) << 4) + ((lane >> 4) << 3) + ((r & 3) << 1);
                afr[mi].u[r] = lds_a[m][kk >> 1];
            }
        }
        for (int ni = 0; ni < 4; ++ni) {
            int n = wn * 64 + ni * 16 + (lane & 15);
            for (int r = 0; r < 8; ++r) {
                int kk = ((r >> 2) << 4) + ((lane >> 4) << 3) + ((r & 3) << 1);
                bfr[ni].u[r] = lds_b[n][kk >> 1];
            }
        }
        for (int mi = 0; mi < 2; ++mi)
            for (int ni = 0; ni < 4; ++ni)
                acc[mi][ni] = __builtin_amdgcn_wmma_f32_16x16x32_bf16(
                    false, afr[mi].v, false, bfr[ni].v,
                    (short)0, acc[mi][ni], false, false);
        __syncthreads();
    }

    for (int mi = 0; mi < 2; ++mi)
        for (int ni = 0; ni < 4; ++ni) {
            int col = ncol0 + wn * 64 + ni * 16 + (lane & 15);
            for (int r = 0; r < 8; ++r) {
                int row = row0 + wm * 32 + mi * 16 + ((lane >> 4) << 3) + r;
                out[(size_t)row * ldo + col] = (__bf16)acc[mi][ni][r];
            }
        }
}

// ---------------------------------------------------------------------------
// Kernel 2: RoPE in place on bf16 Q [BTOK,32,64] and K [BTOK,8,64].
// ---------------------------------------------------------------------------
__global__ __launch_bounds__(256) void rope_kernel(
    __bf16* __restrict__ q, __bf16* __restrict__ k,
    const float* __restrict__ cosb, const float* __restrict__ sinb)
{
    const int PQ = BTOK * N_HEADS * 32;
    const int PK = BTOK * N_KV * 32;
    int idx = blockIdx.x * 256 + threadIdx.x;
    __bf16* base; int d, pos;
    if (idx < PQ) {
        d = idx & 31;
        int h = (idx >> 5) & 31;
        int t = idx >> 10;
        pos = t & (SEQ - 1);
        base = q + ((size_t)t * N_HEADS + h) * HDIM;
    } else {
        idx -= PQ;
        if (idx >= PK) return;
        d = idx & 31;
        int h = (idx >> 5) & 7;
        int t = idx >> 8;
        pos = t & (SEQ - 1);
        base = k + ((size_t)t * N_KV + h) * HDIM;
    }
    float c = cosb[pos * 32 + d], s = sinb[pos * 32 + d];
    float x1 = (float)base[d], x2 = (float)base[d + 32];
    base[d]      = (__bf16)(x1 * c - x2 * s);
    base[d + 32] = (__bf16)(x1 * s + x2 * c);
}

// ---------------------------------------------------------------------------
// Kernel 3: causal FlashAttention (GQA). One block per (b, h, 128-query tile).
// K tile fetched by the Tensor Data Mover (TDM) into LDS; TENSORcnt sync.
// ---------------------------------------------------------------------------
#define BR 128
#define BC 64

__global__ __launch_bounds__(256) void flash_kernel(
    const __bf16* __restrict__ qb,   // [BTOK,32,64]
    const __bf16* __restrict__ kb,   // [BTOK,8,64]
    const __bf16* __restrict__ vb,   // [BTOK,8,64]
    __bf16* __restrict__ ob)         // [BTOK,32,64]
{
    __shared__ unsigned lds_k[BC][32];        // [key][d/2]        (8 KB) -- TDM target
    __shared__ unsigned lds_v[HDIM][32];      // transposed [d][key/2] (8 KB)
    __shared__ unsigned lds_p[8][16][32];     // per wave [row][key/2] (16 KB)

    const int tid = threadIdx.x, wave = tid >> 5, lane = tid & 31;
    int blk = blockIdx.x;
    const int qblk = blk & 15;  blk >>= 4;
    const int h    = blk & 31;  blk >>= 5;
    const int b    = blk;
    const int kvh  = h >> 2;    // N_GROUPS = 4
    const size_t tok0 = (size_t)b * SEQ;
    const unsigned ldsk_off = (unsigned)(uintptr_t)&lds_k[0][0];

    const int qrow0 = qblk * BR + wave * 16;

    // Q fragments (16 rows x 64 d) straight from global (pairs contiguous)
    ABFrag qfr[2];
    {
        const __bf16* qp = qb + (tok0 + qrow0 + (lane & 15)) * (N_HEADS * HDIM)
                              + (size_t)h * HDIM;
        for (int kd = 0; kd < 2; ++kd)
            for (int r = 0; r < 8; ++r) {
                int kk = kd * 32 + ((r >> 2) << 4) + ((lane >> 4) << 3) + ((r & 3) << 1);
                qfr[kd].u[r] = *(const unsigned*)(qp + kk);
            }
    }

    v8f   oacc[4] = {};
    float mrow[8], lrow[8];
    for (int r = 0; r < 8; ++r) { mrow[r] = -__builtin_inff(); lrow[r] = 0.f; }
    const float scale = 0.125f;   // 1/sqrt(64)

    const int nkb = qblk * 2 + 2;   // causal: only key blocks <= diagonal
    for (int kbi = 0; kbi < nkb; ++kbi) {
        // K tile (64 keys x 64 d, bf16) via Tensor Data Mover, wave 0 issues.
        if (wave == 0) {
            const __bf16* kt = kb + (tok0 + kbi * BC) * (N_KV * HDIM)
                                  + (size_t)kvh * HDIM;
            tdm_load_tile_bf16(ldsk_off, kt, /*cols*/HDIM, /*rows*/BC,
                               /*row stride*/N_KV * HDIM);
        }
        // V tile -> LDS transposed [d][key/2] (VALU path; TDM cannot transpose)
        for (int i = 0; i < 8; ++i) {
            int u = tid * 8 + i;
            int d = u >> 5, j = u & 31;
            const __bf16* p = vb + (tok0 + kbi * BC + 2 * j) * (N_KV * HDIM)
                                 + (size_t)kvh * HDIM + d;
            unsigned short lo = __builtin_bit_cast(unsigned short, p[0]);
            unsigned short hi = __builtin_bit_cast(unsigned short, p[N_KV * HDIM]);
            lds_v[d][j] = ((unsigned)hi << 16) | (unsigned)lo;
        }
        if (wave == 0) __builtin_amdgcn_s_wait_tensorcnt(0);
        __syncthreads();

        // Scores: 16 x 64 per wave, 4 N-tiles x 2 K-halves
        float sc[4][8];
        for (int nt = 0; nt < 4; ++nt) {
            v8f c = {};
            for (int kd = 0; kd < 2; ++kd) {
                ABFrag bfr;
                int key = nt * 16 + (lane & 15);
                for (int r = 0; r < 8; ++r) {
                    int dd = kd * 32 + ((r >> 2) << 4) + ((lane >> 4) << 3) + ((r & 3) << 1);
                    bfr.u[r] = lds_k[key][dd >> 1];
                }
                c = __builtin_amdgcn_wmma_f32_16x16x32_bf16(
                        false, qfr[kd].v, false, bfr.v, (short)0, c, false, false);
            }
            int kj = kbi * BC + nt * 16 + (lane & 15);
            for (int r = 0; r < 8; ++r) {
                int qi = qrow0 + ((lane >> 4) << 3) + r;
                sc[nt][r] = (kj <= qi) ? c[r] * scale : -__builtin_inff();
            }
        }

        // Online softmax (row lives across each 16-lane group)
        for (int r = 0; r < 8; ++r) {
            float m = sc[0][r];
            for (int nt = 1; nt < 4; ++nt) m = fmaxf(m, sc[nt][r]);
            for (int off = 1; off < 16; off <<= 1) m = fmaxf(m, __shfl_xor(m, off, 32));
            float mnew = fmaxf(mrow[r], m);
            float alpha = __expf(mrow[r] - mnew);
            mrow[r] = mnew;
            lrow[r] *= alpha;
            for (int dt = 0; dt < 4; ++dt) oacc[dt][r] *= alpha;
            float rsum = 0.f;
            for (int nt = 0; nt < 4; ++nt) {
                float p = __expf(sc[nt][r] - mnew);
                sc[nt][r] = p;
                rsum += p;
            }
            for (int off = 1; off < 16; off <<= 1) rsum += __shfl_xor(rsum, off, 32);
            lrow[r] += rsum;
        }

        // P (C-layout) -> LDS bf16 (A-layout staging)
        for (int nt = 0; nt < 4; ++nt) {
            int col = nt * 16 + (lane & 15);
            for (int r = 0; r < 8; ++r) {
                int row = ((lane >> 4) << 3) + r;
                ((__bf16*)lds_p[wave][row])[col] = (__bf16)sc[nt][r];
            }
        }
        __syncthreads();

        // O += P @ V
        for (int dt = 0; dt < 4; ++dt) {
            for (int kh = 0; kh < 2; ++kh) {
                ABFrag pfr, vfr;
                int m = lane & 15;
                for (int r = 0; r < 8; ++r) {
                    int kk = kh * 32 + ((r >> 2) << 4) + ((lane >> 4) << 3) + ((r & 3) << 1);
                    pfr.u[r] = lds_p[wave][m][kk >> 1];
                }
                int d = dt * 16 + (lane & 15);
                for (int r = 0; r < 8; ++r) {
                    int kk = kh * 32 + ((r >> 2) << 4) + ((lane >> 4) << 3) + ((r & 3) << 1);
                    vfr.u[r] = lds_v[d][kk >> 1];
                }
                oacc[dt] = __builtin_amdgcn_wmma_f32_16x16x32_bf16(
                    false, pfr.v, false, vfr.v, (short)0, oacc[dt], false, false);
            }
        }
        __syncthreads();
    }

    // Normalize and store attn output (bf16, token-major [BTOK, 32, 64])
    for (int dt = 0; dt < 4; ++dt) {
        int d = dt * 16 + (lane & 15);
        for (int r = 0; r < 8; ++r) {
            int qi = qrow0 + ((lane >> 4) << 3) + r;
            float o = oacc[dt][r] / lrow[r];
            ob[((tok0 + qi) * N_HEADS + h) * HDIM + d] = (__bf16)o;
        }
    }
}

// ---------------------------------------------------------------------------
// Kernel 4: output projection. out(f32) = attn(bf16) @ Wo(f32->bf16)
// grid = (32, 16), block = 256
// ---------------------------------------------------------------------------
__global__ __launch_bounds__(256) void out_gemm_kernel(
    const __bf16* __restrict__ A,   // [BTOK, 2048] bf16
    const float*  __restrict__ W,   // [2048, 2048] f32
    float* __restrict__ out)        // [BTOK, 2048] f32
{
    __shared__ unsigned lds_a[BM][BK/2];
    __shared__ unsigned lds_b[BN][BK/2];

    const int tid  = threadIdx.x;
    const int wave = tid >> 5;
    const int lane = tid & 31;
    const int wm   = wave >> 1;
    const int wn   = wave & 1;

    const int row0 = blockIdx.x * BM;
    const int n0   = blockIdx.y * BN;

    v8f acc[2][4] = {};

    for (int k0 = 0; k0 < D_MODEL; k0 += BK) {
        for (int i = 0; i < 8; ++i) {
            int u = tid * 8 + i;
            int m = u >> 4, kp = u & 15;
            const __bf16* p = A + (size_t)(row0 + m) * D_MODEL + k0 + kp * 2;
            lds_a[m][kp] = *(const unsigned*)p;
            __builtin_prefetch(p + BK, 0, 3);
        }
        for (int i = 0; i < 8; ++i) {
            int u = tid * 8 + i;
            int n = u >> 4, kp = u & 15;
            const float* p = W + (size_t)(k0 + kp * 2) * D_MODEL + n0 + n;
            lds_b[n][kp] = pack_bf16(p[0], p[D_MODEL]);
            __builtin_prefetch(p + (size_t)BK * D_MODEL, 0, 3);
        }
        __syncthreads();

        ABFrag afr[2], bfr[4];
        for (int mi = 0; mi < 2; ++mi) {
            int m = wm * 32 + mi * 16 + (lane & 15);
            for (int r = 0; r < 8; ++r) {
                int kk = ((r >> 2) << 4) + ((lane >> 4) << 3) + ((r & 3) << 1);
                afr[mi].u[r] = lds_a[m][kk >> 1];
            }
        }
        for (int ni = 0; ni < 4; ++ni) {
            int n = wn * 64 + ni * 16 + (lane & 15);
            for (int r = 0; r < 8; ++r) {
                int kk = ((r >> 2) << 4) + ((lane >> 4) << 3) + ((r & 3) << 1);
                bfr[ni].u[r] = lds_b[n][kk >> 1];
            }
        }
        for (int mi = 0; mi < 2; ++mi)
            for (int ni = 0; ni < 4; ++ni)
                acc[mi][ni] = __builtin_amdgcn_wmma_f32_16x16x32_bf16(
                    false, afr[mi].v, false, bfr[ni].v,
                    (short)0, acc[mi][ni], false, false);
        __syncthreads();
    }

    for (int mi = 0; mi < 2; ++mi)
        for (int ni = 0; ni < 4; ++ni) {
            int col = n0 + wn * 64 + ni * 16 + (lane & 15);
            for (int r = 0; r < 8; ++r) {
                int row = row0 + wm * 32 + mi * 16 + ((lane >> 4) << 3) + r;
                out[(size_t)row * D_MODEL + col] = acc[mi][ni][r];
            }
        }
}

// ---------------------------------------------------------------------------
extern "C" void kernel_launch(void* const* d_in, const int* in_sizes, int n_in,
                              void* d_out, int out_size, void* d_ws, size_t ws_size,
                              hipStream_t stream) {
    const float* x    = (const float*)d_in[0];
    const float* Wq   = (const float*)d_in[1];
    const float* Wk   = (const float*)d_in[2];
    const float* Wv   = (const float*)d_in[3];
    const float* Wo   = (const float*)d_in[4];
    const float* rcos = (const float*)d_in[5];
    const float* rsin = (const float*)d_in[6];
    float* out = (float*)d_out;

    // Workspace layout (bf16): q[4096x2048] k[4096x512] v[4096x512] attn[4096x2048]
    char* ws = (char*)d_ws;
    const size_t QB = (size_t)BTOK * 2048 * 2;   // 16 MB
    const size_t KB = (size_t)BTOK * 512  * 2;   // 4 MB
    __bf16* q    = (__bf16*)(ws);
    __bf16* k    = (__bf16*)(ws + QB);
    __bf16* v    = (__bf16*)(ws + QB + KB);
    __bf16* attn = (__bf16*)(ws + QB + 2 * KB);

    dim3 g1(BTOK / BM, 3072 / BN);   // 32 x 24
    qkv_gemm_kernel<<<g1, 256, 0, stream>>>(x, Wq, Wk, Wv, q, k, v);

    int rope_threads = BTOK * N_HEADS * 32 + BTOK * N_KV * 32;   // 5,242,880
    rope_kernel<<<(rope_threads + 255) / 256, 256, 0, stream>>>(q, k, rcos, rsin);

    flash_kernel<<<2 * N_HEADS * (SEQ / BR), 256, 0, stream>>>(q, k, v, attn);

    dim3 g2(BTOK / BM, D_MODEL / BN);   // 32 x 16
    out_gemm_kernel<<<g2, 256, 0, stream>>>(attn, Wo, out);
}
